// ImageTransformer_90589450207779
// MI455X (gfx1250) — matrix operand order repaired and checked by reading, not compiled, not run on hardware
//
#include <hip/hip_runtime.h>
#include <cstddef>

// ---------------------------------------------------------------------------
// Problem constants (from the reference)
// ---------------------------------------------------------------------------
constexpr int kB     = 8;
constexpr int kC     = 3;
constexpr int kH     = 32;
constexpr int kW     = 32;
constexpr int kS     = kH * kW * kC;      // 3072 tokens per image
constexpr int kHid   = 512;
constexpr int kHeads = 8;
constexpr int kDh    = kHid / kHeads;     // 64
constexpr int kFilt  = 2048;
constexpr int kL     = 6;
constexpr int kBlen  = 256;
constexpr int kNpix  = 256;
constexpr int kM     = kB * kS;           // 24576 total rows

// ---------------------------------------------------------------------------
// WMMA vector types (CDNA5 wave32) + bf16 helpers
// ---------------------------------------------------------------------------
typedef __attribute__((ext_vector_type(16))) __bf16       v16bf;
typedef __attribute__((ext_vector_type(8)))  float        v8f;
typedef __attribute__((ext_vector_type(4)))  unsigned int v4u;
typedef __attribute__((ext_vector_type(8)))  int          v8i;
typedef __attribute__((ext_vector_type(4)))  int          v4i;

__device__ __forceinline__ unsigned short f2bf(float f) {
  unsigned int u = __float_as_uint(f);
  u = u + 0x7FFFu + ((u >> 16) & 1u);   // round-to-nearest-even
  return (unsigned short)(u >> 16);
}
__device__ __forceinline__ __bf16 bfbits(unsigned short u) {
  union { unsigned short s; __bf16 h; } x; x.s = u; return x.h;
}

// A-matrix (16x32 bf16) per-lane K mapping (ISA 7.12.2): two contiguous
// 8-element runs at K = g*8 and K = 16+g*8 (g = lane/16) -> 2x ds_load_b128.
__device__ __forceinline__ int akmap(int i, int lane) {
  int g = (lane >> 4) & 1;
  return (i < 8) ? (g * 8 + i) : (16 + g * 8 + (i - 8));
}
// B-matrix (32x16 bf16): same K mapping as A (runs at K = g*8 and 16+g*8).
__device__ __forceinline__ int bkmap(int i, int lane) {
  int g = (lane >> 4) & 1;
  return (i < 8) ? (g * 8 + i) : (16 + g * 8 + (i - 8));
}

// ---------------------------------------------------------------------------
// Tensor Data Mover: 2D tile (tile_x elems of 2B per row, tile_y rows,
// row stride = stride_elems) from global -> LDS. D# per ISA ch.8.
// This toolchain exposes the 6-arg builtin:
//   (uint32x4 g0, int32x8 g1, int32x4 g2, int32x4 g3, int32x8 g4, i32 cpol)
// ---------------------------------------------------------------------------
#if __has_builtin(__builtin_amdgcn_tensor_load_to_lds)
#define USE_TDM 1
__device__ __forceinline__ void tdm_load_2d(unsigned lds_off, const void* gptr,
                                            unsigned tile_x, unsigned tile_y,
                                            unsigned long long stride_elems) {
  unsigned long long ga = (unsigned long long)gptr;
  v4u g0;
  g0[0] = 1u;                                   // count=1, user descriptor
  g0[1] = lds_off;                              // lds_addr (bytes)
  g0[2] = (unsigned)(ga & 0xFFFFFFFFu);         // global_addr[31:0]
  g0[3] = (unsigned)((ga >> 32) & 0x1FFFFFFu)   // global_addr[56:32]
          | (2u << 30);                         // type = 2 ("image")
  const unsigned td0 = 1u << 20, td1 = 1u << 20;   // tensor dims (no OOB hit)
  v8i g1;
  g1[0] = (int)(1u << 16);                      // data_size = 1 -> 2 bytes
  g1[1] = (int)((td0 & 0xFFFFu) << 16);         // tensor_dim0[15:0]
  g1[2] = (int)(((td0 >> 16) & 0xFFFFu) | ((td1 & 0xFFFFu) << 16));
  g1[3] = (int)(((td1 >> 16) & 0xFFFFu) | (tile_x << 16));   // tile_dim0
  g1[4] = (int)(tile_y & 0xFFFFu);              // tile_dim1 (tile_dim2 = 0)
  g1[5] = (int)(stride_elems & 0xFFFFFFFFu);    // tensor_dim0_stride lo
  g1[6] = (int)((stride_elems >> 32) & 0xFFFFu);
  g1[7] = 0;
  v4i g2 = {0, 0, 0, 0}, g3 = {0, 0, 0, 0};     // 2D tile: dims 2..4 unused
  v8i g4 = {0, 0, 0, 0, 0, 0, 0, 0};
  __builtin_amdgcn_tensor_load_to_lds(g0, g1, g2, g3, g4, 0);
}
#else
#define USE_TDM 0
#endif

// ---------------------------------------------------------------------------
// Weight convert + transpose: src f32 [Kd][Nd] -> dst bf16 [Nd][Kd].
// ---------------------------------------------------------------------------
__global__ void k_convT(const float* __restrict__ src,
                        unsigned short* __restrict__ dst, int Kd, int Nd) {
  int total = Kd * Nd;
  for (int i = blockIdx.x * blockDim.x + threadIdx.x; i < total;
       i += gridDim.x * blockDim.x) {
    int k = i / Nd, n = i % Nd;
    dst[(size_t)n * Kd + k] = f2bf(src[i]);
  }
}

// ---------------------------------------------------------------------------
// Embedding: gather + sqrt(HID) scale + 1-token shift + sinusoidal positions.
// ---------------------------------------------------------------------------
__global__ void k_embed(const float* __restrict__ X,
                        const float* __restrict__ embeds,
                        float* __restrict__ xf,
                        unsigned short* __restrict__ xb) {
  int tok = blockIdx.x;            // b*S + s
  int b = tok / kS, s = tok % kS;
  int hh = s / (kW * kC), j = s % (kW * kC);

  const float* ebase = nullptr;
  if (s > 0) {                     // shift: token s sources embedding of s-1
    int sp = s - 1;
    int hp = sp / (kW * kC), jp = sp % (kW * kC);
    int c = jp % kC, w = jp / kC;
    float val = X[((size_t)(b * kC + c) * kH + hp) * kW + w];
    int idx = (int)(val * (kNpix - 1));
    idx = idx < 0 ? 0 : (idx > kNpix - 1 ? kNpix - 1 : idx);
    idx += c * kNpix;
    ebase = embeds + (size_t)idx * kHid;
  }
  const float escale = sqrtf((float)kHid);
  const float kinv   = logf(10000.0f) / 127.0f;   // nts-1 = 127

  for (int col = threadIdx.x; col < kHid; col += blockDim.x) {
    float v = ebase ? ebase[col] * escale : 0.0f;
    if (col < 128)      v += sinf((float)hh * expf(-(float)col         * kinv));
    else if (col < 256) v += cosf((float)hh * expf(-(float)(col - 128) * kinv));
    else if (col < 384) v += sinf((float)j  * expf(-(float)(col - 256) * kinv));
    else                v += cosf((float)j  * expf(-(float)(col - 384) * kinv));
    size_t o = (size_t)tok * kHid + col;
    xf[o] = v;
    xb[o] = f2bf(v);
  }
}

// ---------------------------------------------------------------------------
// LayerNorm over HID=512, one block (128 threads) per token.
// ---------------------------------------------------------------------------
__global__ void k_layernorm(const float* __restrict__ in,
                            const float* __restrict__ g,
                            const float* __restrict__ bta,
                            float* __restrict__ xf,
                            unsigned short* __restrict__ xb) {
  __shared__ float red[128], red2[128];
  int tok = blockIdx.x;
  const float* row = in + (size_t)tok * kHid;
  float s = 0.f, s2 = 0.f, vals[4];
#pragma unroll
  for (int e = 0; e < 4; e++) {
    float v = row[threadIdx.x + e * 128];
    vals[e] = v; s += v; s2 += v * v;
  }
  red[threadIdx.x] = s; red2[threadIdx.x] = s2;
  __syncthreads();
  for (int off = 64; off > 0; off >>= 1) {
    if (threadIdx.x < off) {
      red[threadIdx.x]  += red[threadIdx.x + off];
      red2[threadIdx.x] += red2[threadIdx.x + off];
    }
    __syncthreads();
  }
  float mean = red[0] / kHid;
  float var  = red2[0] / kHid - mean * mean;
  float inv  = rsqrtf(var + 1e-6f);
#pragma unroll
  for (int e = 0; e < 4; e++) {
    int col = threadIdx.x + e * 128;
    float y = (vals[e] - mean) * inv * g[col] + bta[col];
    size_t o = (size_t)tok * kHid + col;
    xf[o] = y;
    xb[o] = f2bf(y);
  }
}

// ---------------------------------------------------------------------------
// WMMA GEMM. A[M][K] bf16 row-major, BT[N][K] bf16 (weights pre-transposed).
// Block = 128 threads (4 waves), 64x64 tile, wave = 2x2 WMMA 16x16 tiles,
// K-step 32. Tiles land in LDS via the Tensor Data Mover (wave 0 issues,
// s_wait_tensorcnt, barrier). Both A and B fragments are contiguous
// ds_load_b128 runs. Epilogue fully templated (no runtime branches).
//   VT: store bf16 output transposed to [b,h,d][s] (for attention V).
//   PERM: scatter rows/cols to the (B,C,H,W,NPIX) output layout.
// ---------------------------------------------------------------------------
template <bool BIAS, bool RESID, bool RELU, bool OUTF, bool OUTB, bool PERM,
          bool VT>
__global__ void k_gemm(const unsigned short* __restrict__ A,
                       const unsigned short* __restrict__ BT,
                       const float* __restrict__ bias,
                       const float* __restrict__ resid,
                       float* __restrict__ outF,
                       unsigned short* __restrict__ outB,
                       int M, int N, int K, float alpha) {
  __shared__ unsigned short As[64][32];
  __shared__ unsigned short Bs[64][32];

  const int n0 = blockIdx.x * 64, m0 = blockIdx.y * 64;
  const int tid = threadIdx.x;
  const int lane = tid & 31, wave = tid >> 5;
  const int wm = (wave >> 1) * 32, wn = (wave & 1) * 32;

  v8f zero = {0.f, 0.f, 0.f, 0.f, 0.f, 0.f, 0.f, 0.f};
  v8f acc[2][2];
  acc[0][0] = zero; acc[0][1] = zero; acc[1][0] = zero; acc[1][1] = zero;

#if USE_TDM
  const unsigned ldsA = (unsigned)(unsigned long long)&As[0][0];
  const unsigned ldsB = (unsigned)(unsigned long long)&Bs[0][0];
#endif

  for (int k0 = 0; k0 < K; k0 += 32) {
    __syncthreads();
#if USE_TDM
    if (wave == 0) {
      tdm_load_2d(ldsA, A  + (size_t)m0 * K + k0, 32, 64,
                  (unsigned long long)K);
      tdm_load_2d(ldsB, BT + (size_t)n0 * K + k0, 32, 64,
                  (unsigned long long)K);
      __builtin_amdgcn_s_wait_tensorcnt(0);
    }
#else
#pragma unroll
    for (int e = 0; e < 16; e++) {
      int idx = e * 128 + tid;
      int r = idx >> 5, c = idx & 31;
      As[r][c] = A [(size_t)(m0 + r) * K + k0 + c];
      Bs[r][c] = BT[(size_t)(n0 + r) * K + k0 + c];
    }
#endif
    __syncthreads();

    v16bf af[2], bfr[2];
#pragma unroll
    for (int t = 0; t < 2; t++) {
      int rowA = wm + t * 16 + (lane & 15);
      int rowB = wn + t * 16 + (lane & 15);
#pragma unroll
      for (int i = 0; i < 16; i++) {
        af[t][i]  = bfbits(As[rowA][akmap(i, lane)]);
        bfr[t][i] = bfbits(Bs[rowB][bkmap(i, lane)]);
      }
    }
#pragma unroll
    for (int mi = 0; mi < 2; mi++)
#pragma unroll
      for (int ni = 0; ni < 2; ni++)
        acc[mi][ni] = __builtin_amdgcn_wmma_f32_16x16x32_bf16(
            false, af[mi], false, bfr[ni], (short)0, acc[mi][ni], false, false);
  }

  // Epilogue. C layout: VGPR r -> row = r + 8*(lane>=16), col = lane&15.
#pragma unroll
  for (int mi = 0; mi < 2; mi++)
#pragma unroll
    for (int ni = 0; ni < 2; ni++) {
      int col = n0 + wn + ni * 16 + (lane & 15);
#pragma unroll
      for (int r = 0; r < 8; r++) {
        int row = m0 + wm + mi * 16 + r + ((lane >> 4) << 3);
        float v = acc[mi][ni][r] * alpha;
        if constexpr (BIAS)  v += bias[col];
        if constexpr (RESID) v += resid[(size_t)row * N + col];
        if constexpr (RELU)  v = fmaxf(v, 0.f);
        if constexpr (OUTF) {
          size_t o;
          if constexpr (PERM) {          // (b,s,p) -> (b,c,h,w,p)
            int b = row / kS, s = row % kS;
            int jj = s % (kW * kC), hx = s / (kW * kC);
            int w = jj / kC, c = jj % kC;
            o = ((((size_t)b * kC + c) * kH + hx) * kW + w) * kNpix + col;
          } else {
            o = (size_t)row * N + col;
          }
          outF[o] = v;
        }
        if constexpr (OUTB) {
          if constexpr (VT) {            // V stored [b,h,d][s] for attention
            int b = row / kS, s = row % kS;
            int h = col >> 6, d = col & 63;
            outB[(((size_t)b * kHeads + h) * kDh + d) * kS + s] = f2bf(v);
          } else {
            outB[(size_t)row * N + col] = f2bf(v);
          }
        }
      }
    }
}

// ---------------------------------------------------------------------------
// Block-local causal attention. One wave per (b, head, 16-query tile).
// Q,K in [token][hid] layout (fragment runs contiguous -> global b128 loads);
// V in transposed [b,h,d][s] layout (B fragments contiguous over keys).
// Probabilities staged to LDS as bf16 so P fragments are contiguous too.
// ---------------------------------------------------------------------------
__global__ void k_attn(const unsigned short* __restrict__ Q,
                       const unsigned short* __restrict__ Kt,
                       const unsigned short* __restrict__ VT,
                       unsigned short* __restrict__ O) {
  __shared__ float          sc[16][2 * kBlen];   // logits, 32KB
  __shared__ unsigned short pb[16][2 * kBlen];   // probs (bf16), 16KB

  int gid = blockIdx.x;                 // b*HEADS*(S/16) + h*(S/16) + mt
  int mt = gid % (kS / 16);
  int t  = gid / (kS / 16);
  int h  = t % kHeads;
  int b  = t / kHeads;
  int lane = threadIdx.x;

  int p0 = mt * 16;
  int qb = p0 / kBlen;
  int kstart = (qb == 0) ? 0 : (qb - 1) * kBlen;
  int klen   = (qb == 0) ? kBlen : 2 * kBlen;

  size_t base  = (size_t)b * kS * kHid + (size_t)h * kDh;       // Q/K rows
  size_t baseV = (((size_t)b * kHeads + h) * kDh) * kS;          // VT rows

  // Q fragments (16 rows x dh=64 as two K=32 steps). Q pre-scaled by dh^-0.5.
  v16bf qa[2];
#pragma unroll
  for (int ks = 0; ks < 2; ks++)
#pragma unroll
    for (int i = 0; i < 16; i++) {
      int row = p0 + (lane & 15);
      int d   = ks * 32 + akmap(i, lane);
      qa[ks][i] = bfbits(Q[base + (size_t)row * kHid + d]);
    }

  // Scores: QK^T per 16-key tile, causal mask, stage to LDS.
  for (int kt = 0; kt < klen / 16; kt++) {
    v8f c = {0.f, 0.f, 0.f, 0.f, 0.f, 0.f, 0.f, 0.f};
#pragma unroll
    for (int ks = 0; ks < 2; ks++) {
      v16bf kb;
#pragma unroll
      for (int i = 0; i < 16; i++) {
        int key = kstart + kt * 16 + (lane & 15);
        int d   = ks * 32 + bkmap(i, lane);
        kb[i] = bfbits(Kt[base + (size_t)key * kHid + d]);
      }
      c = __builtin_amdgcn_wmma_f32_16x16x32_bf16(false, qa[ks], false, kb,
                                                  (short)0, c, false, false);
    }
#pragma unroll
    for (int r = 0; r < 8; r++) {
      int row = r + ((lane >> 4) << 3);
      int col = kt * 16 + (lane & 15);
      float v = c[r];
      if (kstart + col > p0 + row) v = -1e9f;   // causal mask
      sc[row][col] = v;
    }
  }
  __syncthreads();

  // Row softmax (lanes 0..15, one query row each); write bf16 probs.
  if (lane < 16) {
    float m = -1e30f;
    for (int kk = 0; kk < klen; kk++) m = fmaxf(m, sc[lane][kk]);
    float sum = 0.f;
    for (int kk = 0; kk < klen; kk++) {
      float e = __expf(sc[lane][kk] - m);
      sc[lane][kk] = e;
      sum += e;
    }
    float inv = 1.0f / sum;
    for (int kk = 0; kk < klen; kk++) pb[lane][kk] = f2bf(sc[lane][kk] * inv);
  }
  __syncthreads();

  // O = P @ V : A = probs (16 x klen) from LDS, B = V^T columns (contiguous).
  v8f o[4];
  o[0] = o[1] = o[2] = o[3] = (v8f){0.f, 0.f, 0.f, 0.f, 0.f, 0.f, 0.f, 0.f};
  for (int kt = 0; kt < klen / 32; kt++) {
    v16bf pa;
#pragma unroll
    for (int i = 0; i < 16; i++)
      pa[i] = bfbits(pb[lane & 15][kt * 32 + akmap(i, lane)]);
#pragma unroll
    for (int nt = 0; nt < 4; nt++) {
      v16bf vb;
#pragma unroll
      for (int i = 0; i < 16; i++) {
        int d   = nt * 16 + (lane & 15);
        int key = kstart + kt * 32 + bkmap(i, lane);
        vb[i] = bfbits(VT[baseV + (size_t)d * kS + key]);
      }
      o[nt] = __builtin_amdgcn_wmma_f32_16x16x32_bf16(false, pa, false, vb,
                                                      (short)0, o[nt], false,
                                                      false);
    }
  }
#pragma unroll
  for (int nt = 0; nt < 4; nt++)
#pragma unroll
    for (int r = 0; r < 8; r++) {
      int row = p0 + r + ((lane >> 4) << 3);
      int d   = nt * 16 + (lane & 15);
      O[base + (size_t)row * kHid + d] = f2bf(o[nt][r]);
    }
}

// ---------------------------------------------------------------------------
// Host orchestration
// ---------------------------------------------------------------------------
extern "C" void kernel_launch(void* const* d_in, const int* in_sizes, int n_in,
                              void* d_out, int out_size, void* d_ws,
                              size_t ws_size, hipStream_t stream) {
  (void)in_sizes; (void)n_in; (void)out_size; (void)ws_size;

  const float* X      = (const float*)d_in[0];
  const float* embeds = (const float*)d_in[1];
  const float* Wq     = (const float*)d_in[2];
  const float* Wk     = (const float*)d_in[3];
  const float* Wv     = (const float*)d_in[4];
  const float* Wo     = (const float*)d_in[5];
  const float* ln1_g  = (const float*)d_in[6];
  const float* ln1_b  = (const float*)d_in[7];
  const float* W1     = (const float*)d_in[8];
  const float* b1     = (const float*)d_in[9];
  const float* W2     = (const float*)d_in[10];
  const float* b2     = (const float*)d_in[11];
  const float* ln2_g  = (const float*)d_in[12];
  const float* ln2_b  = (const float*)d_in[13];
  const float* Wout   = (const float*)d_in[14];
  const float* bout   = (const float*)d_in[15];
  float* out = (float*)d_out;

  char* p = (char*)d_ws;
  auto take = [&](size_t bytes) -> void* {
    void* r = (void*)p;
    p += (bytes + 255) & ~(size_t)255;
    return r;
  };
  float*          xf   = (float*)take((size_t)kM * kHid * 4);
  unsigned short* xb   = (unsigned short*)take((size_t)kM * kHid * 2);
  float*          tmpf = (float*)take((size_t)kM * kHid * 4);
  unsigned short* qbuf = (unsigned short*)take((size_t)kM * kHid * 2);
  unsigned short* kbuf = (unsigned short*)take((size_t)kM * kHid * 2);
  unsigned short* vbuf = (unsigned short*)take((size_t)kM * kHid * 2);
  unsigned short* abuf = (unsigned short*)take((size_t)kM * kHid * 2);
  unsigned short* hbuf = (unsigned short*)take((size_t)kM * kFilt * 2);
  unsigned short* wqb  = (unsigned short*)take((size_t)kHid * kHid * 2);
  unsigned short* wkb  = (unsigned short*)take((size_t)kHid * kHid * 2);
  unsigned short* wvb  = (unsigned short*)take((size_t)kHid * kHid * 2);
  unsigned short* wob  = (unsigned short*)take((size_t)kHid * kHid * 2);
  unsigned short* w1b  = (unsigned short*)take((size_t)kHid * kFilt * 2);
  unsigned short* w2b  = (unsigned short*)take((size_t)kFilt * kHid * 2);
  unsigned short* woutb= (unsigned short*)take((size_t)kHid * kNpix * 2);

  // Convert+transpose weights: f32 [K][N] -> bf16 [N][K].
  auto convT = [&](const float* src, unsigned short* dst, int Kd, int Nd) {
    int n = Kd * Nd;
    int blocks = (n + 255) / 256;
    if (blocks > 8192) blocks = 8192;
    k_convT<<<dim3(blocks), dim3(256), 0, stream>>>(src, dst, Kd, Nd);
  };

  // Front-end.
  k_embed<<<dim3(kM), dim3(128), 0, stream>>>(X, embeds, xf, xb);
  convT(Wout, woutb, kHid, kNpix);

  const float scale = 1.0f / sqrtf((float)kDh);   // dh^-0.5 fused into Q
  const dim3 blk(128);
  const dim3 gridHid(kHid / 64, kM / 64);
  const dim3 gridFilt(kFilt / 64, kM / 64);
  const dim3 gridOut(kNpix / 64, kM / 64);

  for (int i = 0; i < kL; i++) {
    convT(Wq + (size_t)i * kHid * kHid, wqb, kHid, kHid);
    convT(Wk + (size_t)i * kHid * kHid, wkb, kHid, kHid);
    convT(Wv + (size_t)i * kHid * kHid, wvb, kHid, kHid);
    convT(Wo + (size_t)i * kHid * kHid, wob, kHid, kHid);
    convT(W1 + (size_t)i * kHid * kFilt, w1b, kHid, kFilt);
    convT(W2 + (size_t)i * kFilt * kHid, w2b, kFilt, kHid);

    // QKV projections (bf16 out; V stored transposed for attention).
    k_gemm<false, false, false, false, true, false, false>
        <<<gridHid, blk, 0, stream>>>(xb, wqb, nullptr, nullptr, nullptr, qbuf,
                                      kM, kHid, kHid, scale);
    k_gemm<false, false, false, false, true, false, false>
        <<<gridHid, blk, 0, stream>>>(xb, wkb, nullptr, nullptr, nullptr, kbuf,
                                      kM, kHid, kHid, 1.f);
    k_gemm<false, false, false, false, true, false, true>
        <<<gridHid, blk, 0, stream>>>(xb, wvb, nullptr, nullptr, nullptr, vbuf,
                                      kM, kHid, kHid, 1.f);

    // Block-local attention.
    k_attn<<<dim3(kB * kHeads * (kS / 16)), dim3(32), 0, stream>>>(qbuf, kbuf,
                                                                   vbuf, abuf);

    // Output projection + residual, then LN1.
    k_gemm<false, true, false, true, false, false, false>
        <<<gridHid, blk, 0, stream>>>(abuf, wob, nullptr, xf, tmpf, nullptr,
                                      kM, kHid, kHid, 1.f);
    k_layernorm<<<dim3(kM), dim3(128), 0, stream>>>(
        tmpf, ln1_g + (size_t)i * kHid, ln1_b + (size_t)i * kHid, xf, xb);

    // FFN: relu(x@W1+b1) -> bf16 hidden; h@W2+b2 + residual; LN2.
    k_gemm<true, false, true, false, true, false, false>
        <<<gridFilt, blk, 0, stream>>>(xb, w1b, b1 + (size_t)i * kFilt,
                                       nullptr, nullptr, hbuf, kM, kFilt, kHid,
                                       1.f);
    k_gemm<true, true, false, true, false, false, false>
        <<<gridHid, blk, 0, stream>>>(hbuf, w2b, b2 + (size_t)i * kHid, xf,
                                      tmpf, nullptr, kM, kHid, kFilt, 1.f);
    k_layernorm<<<dim3(kM), dim3(128), 0, stream>>>(
        tmpf, ln2_g + (size_t)i * kHid, ln2_b + (size_t)i * kHid, xf, xb);
  }

  // Output projection with fused (B,S,NPIX) -> (B,C,H,W,NPIX) permute.
  k_gemm<true, false, false, true, false, true, false>
      <<<gridOut, blk, 0, stream>>>(xb, woutb, bout, nullptr, out, nullptr,
                                    kM, kNpix, kHid, 1.f);
}
